// GATLayoutModel_68178310856859
// MI455X (gfx1250) — compile-verified, matrix operand order checked by reading, not run on hardware
//
#include <hip/hip_runtime.h>
#include <math.h>

// ---------------------------------------------------------------------------
// GATLayoutModel for MI455X (gfx1250), wave32.
//  - all GEMMs (incl. both channel-attention linears) via v_wmma_f32_16x16x32_bf16
//  - LDS tile staging via TDM tensor_load_to_lds (+ s_wait_tensorcnt) when the
//    builtin exists; uint4 copy fallback otherwise
//  - graph_norm over C=16 folds into one WMMA M-tile: lane sum + shfl_xor(16)
//  - epilogue params staged in LDS; outputs written through LDS, stored b128
//
// Input index map (top-level dict insertion order; params flattened JAX-style:
// dict keys sorted, lists in order):
//   0 x_node_cfg   1 x_feat   2 x_op   3 edge_index
//   4 attn0.l1.b 5 attn0.l1.w 6 attn0.l2.b 7 attn0.l2.w
//   8 clf.b 9 clf.w
//   10+10*i convs[i]: attn.l1.b, attn.l1.w, attn.l2.b, attn.l2.w,
//                     norm.bias, norm.mean_scale, norm.weight,
//                     sage.lin_l.b, sage.lin_l.w, sage.w_r
//   40..43 dattn1   44..47 dattn2   48,49 dense1   50,51 dense2
//   52 emb_layout 53 emb_op 54 lin0.b 55 lin0.w 56 norm0.bias 57 norm0.mean_scale 58 norm0.weight
// ---------------------------------------------------------------------------

#define CC    16
#define NN    10000
#define EE    20000
#define FF    128
#define NROWS (NN * CC)
#define KPAD  224
#define EPSF  1e-5f

typedef __bf16 v16bf __attribute__((ext_vector_type(16)));
typedef float  v8f   __attribute__((ext_vector_type(8)));
typedef unsigned int u32x4 __attribute__((ext_vector_type(4)));
typedef int          i32x4 __attribute__((ext_vector_type(4)));
typedef int          i32x8 __attribute__((ext_vector_type(8)));

#if defined(__has_builtin)
#  if __has_builtin(__builtin_amdgcn_tensor_load_to_lds) && \
      __has_builtin(__builtin_amdgcn_s_wait_tensorcnt)
#    define HAVE_TDM 1
#  endif
#endif
#ifndef HAVE_TDM
#  define HAVE_TDM 0
#endif

// --- 1-D TDM copy: contiguous nbytes (multiple of 8) global -> LDS ---------
__device__ __forceinline__ void tdm_load_1d(const void* g, void* l, unsigned nbytes) {
#if HAVE_TDM
  unsigned long long ga = (unsigned long long)g;
  unsigned la = (unsigned)(unsigned long long)l;
  unsigned n8 = nbytes >> 3;
  u32x4 g0 = {0u, 0u, 0u, 0u};
  i32x8 g1 = {0, 0, 0, 0, 0, 0, 0, 0};
  i32x4 z4 = {0, 0, 0, 0};
  g0.x = 1u;                                                // count=1, user D#
  g0.y = la;                                                // lds_addr
  g0.z = (unsigned)ga;                                      // global_addr[31:0]
  g0.w = (unsigned)((ga >> 32) & 0x01FFFFFFu) | (2u << 30); // ga[56:32] | type=2
  g1[0] = (int)(3u << 16);                                  // data_size = 8 bytes
  g1[1] = (int)((n8 & 0xFFFFu) << 16);                      // tensor_dim0[15:0]
  g1[2] = (int)(((n8 >> 16) & 0xFFFFu) | (1u << 16));       // dim0[31:16], dim1=1
  g1[3] = (int)((n8 & 0xFFFFu) << 16);                      // tile_dim0
  g1[5] = (int)n8;                                          // tensor_dim0_stride
#if __clang_major__ >= 23
  i32x8 z8 = {0, 0, 0, 0, 0, 0, 0, 0};
  __builtin_amdgcn_tensor_load_to_lds(g0, g1, z4, z4, z8, 0);
#else
  __builtin_amdgcn_tensor_load_to_lds(g0, g1, z4, z4, 0);
#endif
#else
  (void)g; (void)l; (void)nbytes;
#endif
}

__device__ __forceinline__ void tdm_wait() {
#if HAVE_TDM
  __builtin_amdgcn_s_wait_tensorcnt(0);
#endif
}

__device__ __forceinline__ void lds_copy16(void* dst, const void* src, int nbytes, int tid) {
  const uint4* s = (const uint4*)src;
  uint4* d = (uint4*)dst;
  for (int i = tid; i < nbytes / 16; i += 256) d[i] = s[i];
}

// --- WMMA fragment loader (A 16xK / B Kx16, 16-bit; ISA 7.12.2 layout) -----
__device__ __forceinline__ v16bf frag16(const __bf16* base, int ldk, int lane, int kt) {
  const __bf16* p = base + (lane & 15) * ldk + kt * 32 + (lane >> 4) * 8;
  v16bf a;
#pragma unroll
  for (int j = 0; j < 8; ++j) a[j] = p[j];
#pragma unroll
  for (int j = 0; j < 8; ++j) a[8 + j] = p[16 + j];
  return a;
}

// 8-wide N-tile sweep sharing one A fragment; reuse-A hint on first 7
// (reuse flag must be a frontend constant -> literal per step).
#define WMMA_NT(NT, RA, WB, LDK)                                              \
  {                                                                           \
    v16bf b = frag16((WB) + (NT) * 16 * (LDK), (LDK), lane, kt);              \
    acc[NT] = __builtin_amdgcn_wmma_f32_16x16x32_bf16(                        \
        false, a, false, b, (short)0, acc[NT], RA, false);                    \
  }
#define WMMA_ROW(WB, LDK)                                                     \
  WMMA_NT(0, true, WB, LDK) WMMA_NT(1, true, WB, LDK)                         \
  WMMA_NT(2, true, WB, LDK) WMMA_NT(3, true, WB, LDK)                         \
  WMMA_NT(4, true, WB, LDK) WMMA_NT(5, true, WB, LDK)                         \
  WMMA_NT(6, true, WB, LDK) WMMA_NT(7, false, WB, LDK)

__device__ __forceinline__ float sigm(float x) { return 1.0f / (1.0f + __expf(-x)); }

// --- utility kernels -------------------------------------------------------
__global__ void zero_f32(float* p, int n) {
  int i = blockIdx.x * 256 + threadIdx.x;
  if (i < n) p[i] = 0.0f;
}

__global__ void cvt_bf16(__bf16* dst, const float* src, int n) {
  int i = blockIdx.x * 256 + threadIdx.x;
  if (i < n) dst[i] = (__bf16)src[i];
}

__global__ void padcvt_bf16(__bf16* dst, const float* src, int rows, int cols, int colspad) {
  int i = blockIdx.x * 256 + threadIdx.x;
  if (i >= rows * colspad) return;
  int r = i / colspad, k = i - r * colspad;
  dst[i] = (__bf16)((k < cols) ? src[r * cols + k] : 0.0f);
}

__global__ void deg_kernel(const int* ei, float* deg) {
  int e = blockIdx.x * 256 + threadIdx.x;
  if (e < EE) atomicAdd(&deg[ei[EE + e]], 1.0f);
}

__global__ void scatter_kernel(const int* ei, const __bf16* X, float* AGG) {
  long long t = (long long)blockIdx.x * 256 + threadIdx.x;
  if (t >= (long long)EE * CC * FF) return;
  int e   = (int)(t >> 11);
  int rem = (int)(t & 2047);
  int src = ei[e], dst = ei[EE + e];
  atomicAdd(&AGG[(long long)dst * 2048 + rem], (float)X[(long long)src * 2048 + rem]);
}

// --- fused lin0 ------------------------------------------------------------
__global__ void lin0_fused(const int* xcfg, const float* xfeat, const int* xop,
                           const float* embL, const float* embO,
                           const __bf16* W0P, const float* b0,
                           const float* n0w, const float* n0b, const float* n0ms,
                           const __bf16* w1b, const float* a1b,
                           const __bf16* w2b, const float* a2b,
                           __bf16* Xout) {
  extern __shared__ char smem[];
  __bf16* sW  = (__bf16*)smem;                    // [128][224]
  __bf16* sA  = sW + FF * KPAD;                   // [128][224]; out tile reuses it
  __bf16* sY  = sA + FF * KPAD;                   // [128][128] normed (bf16)
  __bf16* sW1 = sY + FF * FF;                     // [16][128]
  __bf16* sW2 = sW1 + 16 * FF;                    // [128][32] padded
  __bf16* sH  = sW2 + FF * 32;                    // [128][32] hidden, zero-padded
  float*  sB1 = (float*)(sH + FF * 32);           // [16]
  float*  sB2 = sB1 + 16;                         // [128]
  float*  sPb = sB2 + FF;                         // [128] lin0 bias
  float*  sPm = sPb + FF;                         // [128] mean_scale
  float*  sPw = sPm + FF;                         // [128] weight
  float*  sPn = sPw + FF;                         // [128] norm bias

  const int tid  = threadIdx.x;
  const int row0 = blockIdx.x * 128;
  const int wave = tid >> 5, lane = tid & 31, hi = lane >> 4;

#if HAVE_TDM
  if (tid == 0) {
    tdm_load_1d(W0P, sW, FF * KPAD * 2);
    tdm_load_1d(w1b, sW1, 16 * FF * 2);
    tdm_load_1d(w2b, sW2, FF * 32 * 2);
  }
#else
  lds_copy16(sW, W0P, FF * KPAD * 2, tid);
  lds_copy16(sW1, w1b, 16 * FF * 2, tid);
  lds_copy16(sW2, w2b, FF * 32 * 2, tid);
#endif
  for (int i = tid; i < FF * 32; i += 256) sH[i] = (__bf16)0.0f;
  if (tid < 16)  sB1[tid] = a1b[tid];
  if (tid < 128) {
    sB2[tid] = a2b[tid];
    sPb[tid] = b0[tid]; sPm[tid] = n0ms[tid]; sPw[tid] = n0w[tid]; sPn[tid] = n0b[tid];
  }

  // build this wave's A tile: node n, 16 config rows, concat[feat|cfg|op]
  {
    const int n  = row0 / CC + wave;
    const int c  = lane & 15;
    const int k0 = hi * 112;
    const int op = xop[n];
    __bf16* arow = sA + (wave * 16 + c) * KPAD;
    for (int kk = 0; kk < 112; ++kk) {
      int k = k0 + kk;
      float v;
      if (k < 140)       v = xfeat[n * 140 + k];
      else if (k < 212) { int j = (k - 140) >> 2;
                          int lid = xcfg[((long long)c * NN + n) * 18 + j];
                          v = embL[lid * 4 + ((k - 140) & 3)]; }
      else if (k < 216)  v = embO[op * 4 + (k - 212)];
      else               v = 0.0f;
      arow[k] = (__bf16)v;
    }
  }
#if HAVE_TDM
  if (tid == 0) tdm_wait();
#endif
  __syncthreads();

  // main GEMM 16x128 per wave, K=224 (7*8 = 56 WMMA)
  const __bf16* A = sA + wave * 16 * KPAD;
  v8f acc[8];
#pragma unroll
  for (int nt = 0; nt < 8; ++nt)
#pragma unroll
    for (int j = 0; j < 8; ++j) acc[nt][j] = 0.0f;
#pragma unroll
  for (int kt = 0; kt < 7; ++kt) {
    v16bf a = frag16(A, KPAD, lane, kt);
    WMMA_ROW(sW, KPAD)
  }

  // bias + graph_norm over 16 configs -> sY (bf16); params come from LDS
#pragma unroll
  for (int nt = 0; nt < 8; ++nt) {
    int col = nt * 16 + (lane & 15);
    float bc = sPb[col], ms = sPm[col], wg = sPw[col], bn = sPn[col];
    float s = 0.0f;
#pragma unroll
    for (int j = 0; j < 8; ++j) { acc[nt][j] += bc; s += acc[nt][j]; }
    s += __shfl_xor(s, 16);
    float mean = s * (1.0f / 16.0f);
    float o[8], v = 0.0f;
#pragma unroll
    for (int j = 0; j < 8; ++j) { o[j] = acc[nt][j] - mean * ms; v += o[j] * o[j]; }
    v += __shfl_xor(v, 16);
    float inv = rsqrtf(v * (1.0f / 16.0f) + EPSF);
#pragma unroll
    for (int j = 0; j < 8; ++j)
      sY[(wave * 16 + j + hi * 8) * FF + col] = (__bf16)(wg * o[j] * inv + bn);
  }
  __syncthreads();

  // channel attention via WMMA: H = relu(Y@W1^T+b1); S = H@W2^T+b2
  const __bf16* Yw = sY + wave * 16 * FF;
  v8f a1;
#pragma unroll
  for (int j = 0; j < 8; ++j) a1[j] = 0.0f;
#pragma unroll
  for (int kt = 0; kt < 4; ++kt) {
    v16bf a = frag16(Yw, FF, lane, kt);
    v16bf b = frag16(sW1, FF, lane, kt);
    a1 = __builtin_amdgcn_wmma_f32_16x16x32_bf16(false, a, false, b, (short)0, a1, false, false);
  }
  {
    int jh = lane & 15;
    float b1v = sB1[jh];
#pragma unroll
    for (int j = 0; j < 8; ++j)
      sH[(wave * 16 + j + hi * 8) * 32 + jh] = (__bf16)fmaxf(a1[j] + b1v, 0.0f);
  }
  __bf16* sOut = sA;                              // A tile is dead; reuse as out
  const __bf16* Hw = sH + wave * 16 * 32;
  v16bf aH = frag16(Hw, 32, lane, 0);
#pragma unroll
  for (int nt = 0; nt < 8; ++nt) {
    v8f a2;
#pragma unroll
    for (int j = 0; j < 8; ++j) a2[j] = 0.0f;
    v16bf b = frag16(sW2 + nt * 16 * 32, 32, lane, 0);
    a2 = __builtin_amdgcn_wmma_f32_16x16x32_bf16(false, aH, false, b, (short)0, a2, false, false);
    int f = nt * 16 + (lane & 15);
    float b2v = sB2[f];
#pragma unroll
    for (int j = 0; j < 8; ++j) {
      int m = j + hi * 8;
      float s = sigm(a2[j] + b2v);
      sOut[(wave * 16 + m) * FF + f] = (__bf16)fmaxf(s * (float)Yw[m * FF + f], 0.0f);
    }
  }
  __syncthreads();
  lds_copy16(Xout + (long long)row0 * FF, sOut, FF * FF * 2, tid);  // coalesced b128
}

// --- fused SAGE layer ------------------------------------------------------
__global__ void conv_fused(const __bf16* Xcur, const float* AGG, const float* DEG,
                           const __bf16* Wl, const float* bl, const __bf16* Wr,
                           const float* nw, const float* nb, const float* nms,
                           const __bf16* w1b, const float* a1b,
                           const __bf16* w2b, const float* a2b,
                           __bf16* Xnext) {
  extern __shared__ char smem[];
  __bf16* sWl = (__bf16*)smem;                    // [128][128]
  __bf16* sWr = sWl + FF * FF;
  __bf16* sAg = sWr + FF * FF;                    // agg tile; out tile reuses it
  __bf16* sAx = sAg + FF * FF;
  float*  sT  = (float*)(sAx + FF * FF);          // [128][128] normed sage out
  __bf16* sW1 = (__bf16*)(sT + FF * FF);          // [16][128]
  __bf16* sW2 = sW1 + 16 * FF;                    // [128][32]
  __bf16* sH  = sW2 + FF * 32;                    // [128][32]
  float*  sB1 = (float*)(sH + FF * 32);
  float*  sB2 = sB1 + 16;
  float*  sPb = sB2 + FF;                         // [128] lin_l bias
  float*  sPm = sPb + FF;                         // [128] mean_scale
  float*  sPw = sPm + FF;                         // [128] weight
  float*  sPn = sPw + FF;                         // [128] norm bias
  float*  sSc = sPn + FF;                         // [128] per-row 1/deg

  const int tid  = threadIdx.x;
  const int row0 = blockIdx.x * 128;
  const int wave = tid >> 5, lane = tid & 31, hi = lane >> 4;

#if HAVE_TDM
  if (tid == 0) {
    tdm_load_1d(Wl, sWl, FF * FF * 2);
    tdm_load_1d(Wr, sWr, FF * FF * 2);
    tdm_load_1d(Xcur + (long long)row0 * FF, sAx, FF * FF * 2);
    tdm_load_1d(w1b, sW1, 16 * FF * 2);
    tdm_load_1d(w2b, sW2, FF * 32 * 2);
  }
#else
  lds_copy16(sWl, Wl, FF * FF * 2, tid);
  lds_copy16(sWr, Wr, FF * FF * 2, tid);
  lds_copy16(sAx, Xcur + (long long)row0 * FF, FF * FF * 2, tid);
  lds_copy16(sW1, w1b, 16 * FF * 2, tid);
  lds_copy16(sW2, w2b, FF * 32 * 2, tid);
#endif
  for (int i = tid; i < FF * 32; i += 256) sH[i] = (__bf16)0.0f;
  if (tid < 16)  sB1[tid] = a1b[tid];
  if (tid < 128) {
    sB2[tid] = a2b[tid];
    sPb[tid] = bl[tid]; sPm[tid] = nms[tid]; sPw[tid] = nw[tid]; sPn[tid] = nb[tid];
    sSc[tid] = 1.0f / fmaxf(DEG[(row0 + tid) >> 4], 1.0f);
  }
  __syncthreads();                                 // sSc visible
  // agg tile: scale by 1/deg (from LDS), convert to bf16
  for (int i = tid; i < FF * FF; i += 256) {
    long long R = row0 + (i >> 7);
    sAg[i] = (__bf16)(AGG[R * FF + (i & 127)] * sSc[i >> 7]);
  }
#if HAVE_TDM
  if (tid == 0) tdm_wait();
#endif
  __syncthreads();

  const __bf16* Ag = sAg + wave * 16 * FF;
  const __bf16* Ax = sAx + wave * 16 * FF;
  v8f acc[8];
#pragma unroll
  for (int nt = 0; nt < 8; ++nt)
#pragma unroll
    for (int j = 0; j < 8; ++j) acc[nt][j] = 0.0f;
#pragma unroll
  for (int kt = 0; kt < 4; ++kt) {                 // agg @ lin_l^T
    v16bf a = frag16(Ag, FF, lane, kt);
    WMMA_ROW(sWl, FF)
  }
#pragma unroll
  for (int kt = 0; kt < 4; ++kt) {                 // + x @ w_r^T
    v16bf a = frag16(Ax, FF, lane, kt);
    WMMA_ROW(sWr, FF)
  }

  // bias + graph_norm -> sT (f32); params from LDS
#pragma unroll
  for (int nt = 0; nt < 8; ++nt) {
    int col = nt * 16 + (lane & 15);
    float bc = sPb[col], ms = sPm[col], wg = sPw[col], bn = sPn[col];
    float s = 0.0f;
#pragma unroll
    for (int j = 0; j < 8; ++j) { acc[nt][j] += bc; s += acc[nt][j]; }
    s += __shfl_xor(s, 16);
    float mean = s * (1.0f / 16.0f);
    float o[8], v = 0.0f;
#pragma unroll
    for (int j = 0; j < 8; ++j) { o[j] = acc[nt][j] - mean * ms; v += o[j] * o[j]; }
    v += __shfl_xor(v, 16);
    float inv = rsqrtf(v * (1.0f / 16.0f) + EPSF);
#pragma unroll
    for (int j = 0; j < 8; ++j)
      sT[(wave * 16 + j + hi * 8) * FF + col] = wg * o[j] * inv + bn;
  }
  __syncthreads();

  // channel attention on x via WMMA; out = relu(tmp + sigm(S) * x)
  v8f a1;
#pragma unroll
  for (int j = 0; j < 8; ++j) a1[j] = 0.0f;
#pragma unroll
  for (int kt = 0; kt < 4; ++kt) {
    v16bf a = frag16(Ax, FF, lane, kt);
    v16bf b = frag16(sW1, FF, lane, kt);
    a1 = __builtin_amdgcn_wmma_f32_16x16x32_bf16(false, a, false, b, (short)0, a1, false, false);
  }
  {
    int jh = lane & 15;
    float b1v = sB1[jh];
#pragma unroll
    for (int j = 0; j < 8; ++j)
      sH[(wave * 16 + j + hi * 8) * 32 + jh] = (__bf16)fmaxf(a1[j] + b1v, 0.0f);
  }
  __bf16* sOut = sAg;                              // agg tile dead; reuse as out
  const __bf16* Hw = sH + wave * 16 * 32;
  v16bf aH = frag16(Hw, 32, lane, 0);
#pragma unroll
  for (int nt = 0; nt < 8; ++nt) {
    v8f a2;
#pragma unroll
    for (int j = 0; j < 8; ++j) a2[j] = 0.0f;
    v16bf b = frag16(sW2 + nt * 16 * 32, 32, lane, 0);
    a2 = __builtin_amdgcn_wmma_f32_16x16x32_bf16(false, aH, false, b, (short)0, a2, false, false);
    int f = nt * 16 + (lane & 15);
    float b2v = sB2[f];
#pragma unroll
    for (int j = 0; j < 8; ++j) {
      int m = j + hi * 8, row = wave * 16 + m;
      float s = sigm(a2[j] + b2v);
      sOut[row * FF + f] = (__bf16)fmaxf(sT[row * FF + f] + s * (float)Ax[m * FF + f], 0.0f);
    }
  }
  __syncthreads();
  lds_copy16(Xnext + (long long)row0 * FF, sOut, FF * FF * 2, tid);  // coalesced b128
}

// --- mean over nodes -------------------------------------------------------
__global__ void pool_kernel(const __bf16* X, float* POOL) {
  int idx = blockIdx.x * 256 + threadIdx.x;
  if (idx >= CC * FF) return;
  int c = idx >> 7, f = idx & 127;
  float s = 0.0f;
  for (int n = 0; n < NN; ++n) {
    if ((n & 63) == 0)
      __builtin_prefetch(&X[((long long)(n + 64) * CC + c) * FF + f], 0, 1);
    s += (float)X[((long long)n * CC + c) * FF + f];
  }
  POOL[idx] = s * (1.0f / NN);
}

// --- tiny head (per-row state in LDS to avoid scratch spills) --------------
__global__ void head_kernel(const float* POOL,
                            const float* d1w, const float* d1b,
                            const float* a1w, const float* a1b, const float* a2w, const float* a2b,
                            const float* d2w, const float* d2b,
                            const float* b1w, const float* b1b, const float* b2w, const float* b2b,
                            const float* cw, const float* cb, float* out) {
  __shared__ float zs[CC][FF];
  __shared__ float z2s[CC][FF];
  __shared__ float rows[CC][FF];
  int c = threadIdx.x;
  if (c < CC) {
    const float* x0 = POOL + c * FF;
    for (int o = 0; o < FF; ++o) {
      float t = d1b[o];
      for (int f = 0; f < FF; ++f) t += x0[f] * d1w[o * FF + f];
      zs[c][o] = t;
    }
    float h[16];
    for (int j = 0; j < 16; ++j) {
      float t = a1b[j];
      for (int o = 0; o < FF; ++o) t += zs[c][o] * a1w[j * FF + o];
      h[j] = fmaxf(t, 0.0f);
    }
    for (int o = 0; o < FF; ++o) {
      float t = a2b[o];
      for (int j = 0; j < 16; ++j) t += h[j] * a2w[o * 16 + j];
      zs[c][o] = fmaxf(sigm(t) * zs[c][o], 0.0f);
    }
    for (int o = 0; o < FF; ++o) {
      float t = d2b[o];
      for (int f = 0; f < FF; ++f) t += zs[c][f] * d2w[o * FF + f];
      z2s[c][o] = t;
    }
    for (int j = 0; j < 16; ++j) {
      float t = b1b[j];
      for (int o = 0; o < FF; ++o) t += z2s[c][o] * b1w[j * FF + o];
      h[j] = fmaxf(t, 0.0f);
    }
    for (int o = 0; o < FF; ++o) {
      float t = b2b[o];
      for (int j = 0; j < 16; ++j) t += h[j] * b2w[o * 16 + j];
      rows[c][o] = fmaxf(sigm(t) * z2s[c][o], 0.0f);
    }
  }
  __syncthreads();
  if (c < CC) {
    float r = cb[0];
    for (int f = 0; f < FF; ++f) {
      float m = 0.0f;
      for (int cc2 = 0; cc2 < CC; ++cc2) m += rows[cc2][f];
      r += (rows[c][f] - m * (1.0f / CC)) * cw[f];
    }
    out[c] = r;
  }
}

// ---------------------------------------------------------------------------
extern "C" void kernel_launch(void* const* d_in, const int* in_sizes, int n_in,
                              void* d_out, int out_size, void* d_ws, size_t ws_size,
                              hipStream_t stream) {
  (void)in_sizes; (void)n_in; (void)out_size; (void)ws_size;
  const int*   xcfg = (const int*)  d_in[0];
  const float* xfeat= (const float*)d_in[1];
  const int*   xop  = (const int*)  d_in[2];
  const int*   ei   = (const int*)  d_in[3];
  const float* a0l1b=(const float*)d_in[4],  *a0l1w=(const float*)d_in[5];
  const float* a0l2b=(const float*)d_in[6],  *a0l2w=(const float*)d_in[7];
  const float* clfb =(const float*)d_in[8],  *clfw =(const float*)d_in[9];
  const float* da1l1b=(const float*)d_in[40], *da1l1w=(const float*)d_in[41];
  const float* da1l2b=(const float*)d_in[42], *da1l2w=(const float*)d_in[43];
  const float* da2l1b=(const float*)d_in[44], *da2l1w=(const float*)d_in[45];
  const float* da2l2b=(const float*)d_in[46], *da2l2w=(const float*)d_in[47];
  const float* d1b=(const float*)d_in[48], *d1w=(const float*)d_in[49];
  const float* d2b=(const float*)d_in[50], *d2w=(const float*)d_in[51];
  const float* embL=(const float*)d_in[52], *embO=(const float*)d_in[53];
  const float* l0b =(const float*)d_in[54], *l0w =(const float*)d_in[55];
  const float* n0b =(const float*)d_in[56], *n0ms=(const float*)d_in[57], *n0w=(const float*)d_in[58];

  // workspace carve
  __bf16* X0  = (__bf16*)d_ws;
  __bf16* X1  = X0 + (long long)NROWS * FF;
  float*  AGG = (float*)(X1 + (long long)NROWS * FF);
  float*  DEG = AGG + (long long)NROWS * FF;
  float*  POOL= DEG + NN;
  __bf16* W0P = (__bf16*)(POOL + CC * FF);        // 128*224
  __bf16* WLB = W0P + FF * KPAD;                  // 3 * 16384
  __bf16* WRB = WLB + 3 * FF * FF;
  __bf16* AW1 = WRB + 3 * FF * FF;                // 4 * 2048 (attn0, conv0..2)
  __bf16* AW2 = AW1 + 4 * 16 * FF;                // 4 * 4096 (128x32 padded)

  const size_t lds_lin0 = (size_t)(2 * FF * KPAD + FF * FF + 16 * FF + 2 * FF * 32) * 2
                        + (16 + 128 + 4 * 128) * 4;
  const size_t lds_conv = (size_t)(4 * FF * FF + 16 * FF + 2 * FF * 32) * 2
                        + (size_t)FF * FF * 4 + (16 + 128 + 5 * 128) * 4;

  // --- weight pre-conversion to bf16 (also builds zero-padded layouts) ---
  padcvt_bf16<<<(FF * KPAD + 255) / 256, 256, 0, stream>>>(W0P, l0w, FF, 216, KPAD);
  cvt_bf16<<<(16 * FF + 255) / 256, 256, 0, stream>>>(AW1, a0l1w, 16 * FF);
  padcvt_bf16<<<(FF * 32 + 255) / 256, 256, 0, stream>>>(AW2, a0l2w, FF, 16, 32);
  for (int L = 0; L < 3; ++L) {
    const float* cl1w=(const float*)d_in[10 + 10 * L + 1];
    const float* cl2w=(const float*)d_in[10 + 10 * L + 3];
    const float* slw =(const float*)d_in[10 + 10 * L + 8];
    const float* swr =(const float*)d_in[10 + 10 * L + 9];
    cvt_bf16<<<(FF * FF + 255) / 256, 256, 0, stream>>>(WLB + L * FF * FF, slw, FF * FF);
    cvt_bf16<<<(FF * FF + 255) / 256, 256, 0, stream>>>(WRB + L * FF * FF, swr, FF * FF);
    cvt_bf16<<<(16 * FF + 255) / 256, 256, 0, stream>>>(AW1 + (1 + L) * 16 * FF, cl1w, 16 * FF);
    padcvt_bf16<<<(FF * 32 + 255) / 256, 256, 0, stream>>>(AW2 + (1 + L) * FF * 32, cl2w, FF, 16, 32);
  }

  zero_f32<<<(NN + 255) / 256, 256, 0, stream>>>(DEG, NN);
  deg_kernel<<<(EE + 255) / 256, 256, 0, stream>>>(ei, DEG);

  lin0_fused<<<NN / 8, 256, lds_lin0, stream>>>(
      xcfg, xfeat, xop, embL, embO, W0P, l0b, n0w, n0b, n0ms,
      AW1, a0l1b, AW2, a0l2b, X0);

  const long long scatN = (long long)EE * CC * FF;
  __bf16* Xc = X0; __bf16* Xn = X1;
  for (int L = 0; L < 3; ++L) {
    const float* cl1b=(const float*)d_in[10 + 10 * L + 0];
    const float* cl2b=(const float*)d_in[10 + 10 * L + 2];
    const float* cnb =(const float*)d_in[10 + 10 * L + 4];
    const float* cnms=(const float*)d_in[10 + 10 * L + 5];
    const float* cnw =(const float*)d_in[10 + 10 * L + 6];
    const float* slb =(const float*)d_in[10 + 10 * L + 7];

    zero_f32<<<(NROWS * FF + 255) / 256, 256, 0, stream>>>(AGG, NROWS * FF);
    scatter_kernel<<<(unsigned)((scatN + 255) / 256), 256, 0, stream>>>(ei, Xc, AGG);
    conv_fused<<<NN / 8, 256, lds_conv, stream>>>(
        Xc, AGG, DEG, WLB + L * FF * FF, slb, WRB + L * FF * FF,
        cnw, cnb, cnms, AW1 + (1 + L) * 16 * FF, cl1b,
        AW2 + (1 + L) * FF * 32, cl2b, Xn);
    __bf16* t = Xc; Xc = Xn; Xn = t;
  }

  pool_kernel<<<(CC * FF + 255) / 256, 256, 0, stream>>>(Xc, POOL);
  head_kernel<<<1, 64, 0, stream>>>(POOL, d1w, d1b,
      da1l1w, da1l1b, da1l2w, da1l2b, d2w, d2b,
      da2l1w, da2l1b, da2l2w, da2l2b, clfw, clfb, (float*)d_out);
}
